// GCNNet_14551349199045
// MI455X (gfx1250) — matrix operand order repaired and checked by reading, not compiled, not run on hardware
//
#include <hip/hip_runtime.h>
#include <hip/hip_bf16.h>

#define N_NODES 100000
#define N_EDGES 1600000

typedef __attribute__((ext_vector_type(16))) __bf16 v16bf;
typedef __attribute__((ext_vector_type(8)))  float  v8f;

// ---------------------------------------------------------------- utilities
__global__ void zero_f32_kernel(float* __restrict__ p, long n) {
  long i = (long)blockIdx.x * blockDim.x + threadIdx.x;
  if (i < n) p[i] = 0.0f;
}

__global__ void fill_f32_kernel(float* __restrict__ p, float v, int n) {
  int i = blockIdx.x * blockDim.x + threadIdx.x;
  if (i < n) p[i] = v;
}

// deg[dst] += 1 per edge (self-loop contribution pre-seeded as 1.0)
__global__ void deg_accum_kernel(const int* __restrict__ dst, float* __restrict__ deg, int e) {
  int i = blockIdx.x * blockDim.x + threadIdx.x;
  if (i < e) atomicAdd(&deg[dst[i]], 1.0f);
}

__global__ void rsqrt_inplace_kernel(float* __restrict__ d, int n) {
  int i = blockIdx.x * blockDim.x + threadIdx.x;
  if (i < n) d[i] = rsqrtf(d[i]);  // deg >= 1 always (self loops)
}

// per-edge symmetric norm: norm[e] = dinv[src[e]] * dinv[dst[e]]
__global__ void edge_norm_kernel(const int* __restrict__ src, const int* __restrict__ dst,
                                 const float* __restrict__ dinv, float* __restrict__ nrm, int e) {
  int i = blockIdx.x * blockDim.x + threadIdx.x;
  if (i < e) nrm[i] = dinv[src[i]] * dinv[dst[i]];
}

// ------------------------------------------------- pack W into WMMA B-fragments
// B (bf16, 32x16 per k-chunk): lane 0-15 -> N=lane, K={0..7,16..23} (2/VGPR);
// lane 16-31 -> N=lane-16, K={8..15,24..31}.  Layout: pack[tile][kchunk][lane][16]
__global__ void pack_w_kernel(const float* __restrict__ W, __bf16* __restrict__ pack,
                              int Nfull, int ntiles, int kchunks) {
  int idx = blockIdx.x * blockDim.x + threadIdx.x;
  int total = ntiles * kchunks * 32;
  if (idx >= total) return;
  int tile = idx / (kchunks * 32);
  int rem  = idx - tile * (kchunks * 32);
  int kc   = rem >> 5;
  int lane = rem & 31;
  int n  = tile * 16 + (lane & 15);
  int kb = (lane < 16) ? 0 : 16;
  __bf16* dstp = pack + (size_t)idx * 16;
#pragma unroll
  for (int i = 0; i < 16; ++i) {
    int k = kc * 32 + ((i < 8) ? (kb >> 1) + i : 16 + (kb >> 1) + (i - 8));
    float v = (n < Nfull) ? W[(size_t)k * Nfull + n] : 0.0f;
    dstp[i] = (__bf16)v;
  }
}

// ------------------------------------------------- WMMA GEMM: C = A(fp32) * Bpack(bf16)
// One wave computes a full 16 x (16*NTILES) output slab: A fragment is loaded
// and converted fp32->bf16 ONCE per k-chunk, then reused across all N-tiles
// (register-blocked accumulators, NTILES x v8f).  A-layout per ISA 7.12.2:
// lane<16 -> M=lane&15, K in {klo..klo+7, 16+klo..16+klo+7}, klo=(lane>=16)*8.
template <int KCHUNKS, int NTILES>
__global__ __launch_bounds__(32)
void gcn_gemm_wmma(const float* __restrict__ A, const __bf16* __restrict__ Bp,
                   float* __restrict__ C, int lda, int ldc) {
  const int lane = threadIdx.x;
  const int m0   = blockIdx.x * 16;
  const int arow = m0 + (lane & 15);
  const int klo  = (lane >> 4) << 3;
  const float* aptr = A + (size_t)arow * lda + klo;
  const v16bf* bbase = (const v16bf*)Bp + lane;

  v8f acc[NTILES];
#pragma unroll
  for (int t = 0; t < NTILES; ++t) acc[t] = (v8f){};

#pragma unroll
  for (int kc = 0; kc < KCHUNKS; ++kc) {
    const float4 f0 = *(const float4*)(aptr + kc * 32);
    const float4 f1 = *(const float4*)(aptr + kc * 32 + 4);
    const float4 f2 = *(const float4*)(aptr + kc * 32 + 16);
    const float4 f3 = *(const float4*)(aptr + kc * 32 + 20);
    v16bf a;
    a[0]  = (__bf16)f0.x; a[1]  = (__bf16)f0.y; a[2]  = (__bf16)f0.z; a[3]  = (__bf16)f0.w;
    a[4]  = (__bf16)f1.x; a[5]  = (__bf16)f1.y; a[6]  = (__bf16)f1.z; a[7]  = (__bf16)f1.w;
    a[8]  = (__bf16)f2.x; a[9]  = (__bf16)f2.y; a[10] = (__bf16)f2.z; a[11] = (__bf16)f2.w;
    a[12] = (__bf16)f3.x; a[13] = (__bf16)f3.y; a[14] = (__bf16)f3.z; a[15] = (__bf16)f3.w;
#pragma unroll
    for (int t = 0; t < NTILES; ++t) {
      const v16bf b = bbase[((size_t)t * KCHUNKS + kc) * 32];
      acc[t] = __builtin_amdgcn_wmma_f32_16x16x32_bf16(false, a, false, b, (short)0,
                                                       acc[t], false, false);
    }
  }

  // C/D layout: VGPR r -> M = r + 8*(lane>=16), N = lane&15
  const int crow = m0 + ((lane >> 4) << 3);
  const int c0   = lane & 15;
#pragma unroll
  for (int t = 0; t < NTILES; ++t) {
#pragma unroll
    for (int r = 0; r < 8; ++r) {
      C[(size_t)(crow + r) * ldc + t * 16 + c0] = acc[t][r];
    }
  }
}

// ------------------------------------------------- edge scatter: out[dst] += feat[src]*norm
// One thread per (edge, float4-quad). L2-resident gather + float atomics.
// QUADS templated so edge/quad decomposition is shift/mul, not division.
template <int QUADS>
__global__ void agg_scatter_kernel(const float* __restrict__ feat,
                                   const int* __restrict__ src, const int* __restrict__ dst,
                                   const float* __restrict__ nrm, float* __restrict__ out,
                                   int ldf4, int ldo) {
  long idx = (long)blockIdx.x * blockDim.x + threadIdx.x;
  if (idx >= (long)N_EDGES * QUADS) return;
  int e = (int)(idx / QUADS);
  int q = (int)(idx - (long)e * QUADS);
  int s = src[e];
  int d = dst[e];
  float w = nrm[e];
  const float4 v = ((const float4*)feat)[(size_t)s * ldf4 + q];
  float* o = out + (size_t)d * ldo + q * 4;
  atomicAdd(o + 0, v.x * w);
  atomicAdd(o + 1, v.y * w);
  atomicAdd(o + 2, v.z * w);
  atomicAdd(o + 3, v.w * w);
}

// ------------------------------------------------- layer-1 epilogue: self-loop + bias + ReLU
__global__ void finish_relu_kernel(const float* __restrict__ xw, float* __restrict__ agg,
                                   const float* __restrict__ dinv, const float* __restrict__ bias) {
  long i = (long)blockIdx.x * blockDim.x + threadIdx.x;
  if (i >= (long)N_NODES * 128) return;
  int node = (int)(i >> 7);
  int f    = (int)(i & 127);
  float dn = dinv[node];
  float v  = agg[i] + xw[i] * dn * dn + bias[f];
  agg[i] = v > 0.0f ? v : 0.0f;
}

// ------------------------------------------------- layer-2 epilogue: self-loop + bias + log_softmax
__global__ void final_logsoftmax_kernel(float* __restrict__ out, const float* __restrict__ hw,
                                        const float* __restrict__ dinv, const float* __restrict__ bias) {
  int n = blockIdx.x * blockDim.x + threadIdx.x;
  if (n >= N_NODES) return;
  float dn2 = dinv[n] * dinv[n];
  float r[40];
  float mx = -3.0e38f;
#pragma unroll
  for (int j = 0; j < 40; ++j) {
    float v = out[(size_t)n * 40 + j] + hw[(size_t)n * 48 + j] * dn2 + bias[j];
    r[j] = v;
    mx = fmaxf(mx, v);
  }
  float s = 0.0f;
#pragma unroll
  for (int j = 0; j < 40; ++j) s += expf(r[j] - mx);
  float lse = mx + logf(s);
#pragma unroll
  for (int j = 0; j < 40; ++j) out[(size_t)n * 40 + j] = r[j] - lse;
}

// ---------------------------------------------------------------- launcher
extern "C" void kernel_launch(void* const* d_in, const int* in_sizes, int n_in,
                              void* d_out, int out_size, void* d_ws, size_t ws_size,
                              hipStream_t stream) {
  const float* x   = (const float*)d_in[0];
  const int*  eidx = (const int*)d_in[1];
  const float* W1  = (const float*)d_in[2];
  const float* b1  = (const float*)d_in[3];
  const float* W2  = (const float*)d_in[4];
  const float* b2  = (const float*)d_in[5];
  float* out = (float*)d_out;
  const int* src = eidx;
  const int* dst = eidx + N_EDGES;

  char* ws = (char*)d_ws;
  size_t off = 0;
  auto take = [&](size_t bytes) -> void* {
    void* p = ws + off;
    off = (off + bytes + 255) & ~(size_t)255;
    return p;
  };
  float*  agg1 = (float*)take((size_t)N_NODES * 128 * sizeof(float));   // 51.2 MB
  float*  xw   = (float*)take((size_t)N_NODES * 128 * sizeof(float));   // 51.2 MB (also hw, ld=48)
  float*  dinv = (float*)take((size_t)N_NODES * sizeof(float));         // deg -> dinv in place
  float*  enrm = (float*)take((size_t)N_EDGES * sizeof(float));         // 6.4 MB per-edge norm
  __bf16* w1p  = (__bf16*)take((size_t)8 * 8 * 32 * 16 * sizeof(__bf16));
  __bf16* w2p  = (__bf16*)take((size_t)3 * 4 * 32 * 16 * sizeof(__bf16));

  const int T = 256;

  // zero accumulators (d_out doubles as layer-2 accumulator)
  zero_f32_kernel<<<(int)(((long)N_NODES * 128 + T - 1) / T), T, 0, stream>>>(agg1, (long)N_NODES * 128);
  zero_f32_kernel<<<(int)(((long)N_NODES * 40 + T - 1) / T), T, 0, stream>>>(out, (long)N_NODES * 40);

  // degree (seed 1.0 for self loop) -> rsqrt -> per-edge norm
  fill_f32_kernel<<<(N_NODES + T - 1) / T, T, 0, stream>>>(dinv, 1.0f, N_NODES);
  deg_accum_kernel<<<(N_EDGES + T - 1) / T, T, 0, stream>>>(dst, dinv, N_EDGES);
  rsqrt_inplace_kernel<<<(N_NODES + T - 1) / T, T, 0, stream>>>(dinv, N_NODES);
  edge_norm_kernel<<<(N_EDGES + T - 1) / T, T, 0, stream>>>(src, dst, dinv, enrm, N_EDGES);

  // pack weights into WMMA B-fragment order (W2 padded N 40->48)
  pack_w_kernel<<<(8 * 8 * 32 + T - 1) / T, T, 0, stream>>>(W1, w1p, 128, 8, 8);
  pack_w_kernel<<<(3 * 4 * 32 + T - 1) / T, T, 0, stream>>>(W2, w2p, 40, 3, 4);

  // layer 1: xw = x @ W1   (M=100000, K=256, N=128; one wave per 16x128 slab)
  gcn_gemm_wmma<8, 8><<<N_NODES / 16, 32, 0, stream>>>(x, w1p, xw, 256, 128);
  agg_scatter_kernel<32><<<(int)(((long)N_EDGES * 32 + T - 1) / T), T, 0, stream>>>(
      xw, src, dst, enrm, agg1, 32, 128);
  finish_relu_kernel<<<(int)(((long)N_NODES * 128 + T - 1) / T), T, 0, stream>>>(xw, agg1, dinv, b1);

  // layer 2: hw = h1 @ W2  (M=100000, K=128, N=40 padded to 48); hw reuses xw buffer
  gcn_gemm_wmma<4, 3><<<N_NODES / 16, 32, 0, stream>>>(agg1, w2p, xw, 128, 48);
  agg_scatter_kernel<10><<<(int)(((long)N_EDGES * 10 + T - 1) / T), T, 0, stream>>>(
      xw, src, dst, enrm, out, 12, 40);
  final_logsoftmax_kernel<<<(N_NODES + T - 1) / T, T, 0, stream>>>(out, xw, dinv, b2);

  (void)in_sizes; (void)n_in; (void)out_size; (void)ws_size;
}
